// MemoryEngineLayer_54752243090073
// MI455X (gfx1250) — matrix operand order repaired and checked by reading, not compiled, not run on hardware
//
#include <hip/hip_runtime.h>
#include <cstddef>

#define H     1024
#define SLEN  1024
#define BATCH 16
#define MBASE 256
#define NTOT  272
#define NP    288          // padded slot count (18 tiles of 16)
#define KT_C  (H / 32)     // 32 K-tiles for the c-GEMM
#define KT_Y  (NP / 32)    // 9 K-tiles for the y/resid GEMMs
#define NT_C  (NP / 16)    // 18 N-tiles
#define NT_Y  (H / 16)     // 64 N-tiles

typedef __attribute__((ext_vector_type(16))) _Float16 v16h;
typedef __attribute__((ext_vector_type(8)))  float    v8f;

// CDNA5 WMMA: D = A(16x32 f16) * B(32x16 f16) + C(16x16 f32)
__device__ __forceinline__ v8f wmma16(v16h a, v16h b, v8f c) {
  return __builtin_amdgcn_wmma_f32_16x16x32_f16(
      /*neg_a=*/false, a, /*neg_b=*/false, b,
      /*c_mod=*/(short)0, c, /*reuse_a=*/false, /*reuse_b=*/false);
}

// f16 A-matrix 16x32 layout: element e of lane -> K index
__device__ __forceinline__ int a16k(int e, int hi) {
  return ((e < 8) ? e : e + 8) + hi * 8;
}

// ---------------------------------------------------------------- setup ----

__global__ void k_gram(const float* __restrict__ basis, float* __restrict__ A) {
  int idx = blockIdx.x * blockDim.x + threadIdx.x;
  if (idx >= NP * NP) return;
  int i = idx / NP, j = idx % NP;
  float v;
  if (i < NTOT && j < NTOT) {
    float s = 0.f;
    for (int k = 0; k < H; ++k) s += basis[k * NTOT + i] * basis[k * NTOT + j];
    v = s + ((i == j) ? 1e-3f : 0.f);
  } else {
    v = (i == j) ? 1.f : 0.f;   // identity padding keeps the inverse block-diagonal
  }
  A[idx] = v;
}

__global__ void k_invscale(const float* __restrict__ A, float* __restrict__ invs) {
  __shared__ float mx[256];
  float local = 0.f;
  for (int r = threadIdx.x; r < NP; r += 256) {
    float s = 0.f;
    for (int j = 0; j < NP; ++j) s += fabsf(A[r * NP + j]);
    local = fmaxf(local, s);
  }
  mx[threadIdx.x] = local;
  __syncthreads();
  for (int off = 128; off > 0; off >>= 1) {
    if (threadIdx.x < off) mx[threadIdx.x] = fmaxf(mx[threadIdx.x], mx[threadIdx.x + off]);
    __syncthreads();
  }
  if (threadIdx.x == 0) invs[0] = 1.f / fmaxf(mx[0], 1e-20f);
}

__global__ void k_eyescale(float* __restrict__ X, const float* __restrict__ invs) {
  int idx = blockIdx.x * blockDim.x + threadIdx.x;
  if (idx >= NP * NP) return;
  int i = idx / NP, j = idx % NP;
  X[idx] = (i == j) ? invs[0] : 0.f;
}

__global__ void k_gemmNP(const float* __restrict__ A, const float* __restrict__ Bm,
                         float* __restrict__ C) {
  int idx = blockIdx.x * blockDim.x + threadIdx.x;
  if (idx >= NP * NP) return;
  int i = idx / NP, j = idx % NP;
  float s = 0.f;
  for (int k = 0; k < NP; ++k) s += A[i * NP + k] * Bm[k * NP + j];
  C[idx] = s;
}

__global__ void k_twominus(float* __restrict__ T) {   // T <- 2I - T
  int idx = blockIdx.x * blockDim.x + threadIdx.x;
  if (idx >= NP * NP) return;
  int i = idx / NP, j = idx % NP;
  T[idx] = ((i == j) ? 2.f : 0.f) - T[idx];
}

// Wc[k][n] = sum_j (basis[k][j] + leak[j][k]) * Lc[n][j]
__global__ void k_fold(const float* __restrict__ basis, const float* __restrict__ leak,
                       const float* __restrict__ Lc, float* __restrict__ Wc32) {
  int idx = blockIdx.x * blockDim.x + threadIdx.x;
  if (idx >= H * NP) return;
  int k = idx / NP, n = idx % NP;
  float s = 0.f;
  if (n < NTOT) {
    for (int j = 0; j < NTOT; ++j)
      s += (basis[k * NTOT + j] + leak[j * H + k]) * Lc[n * NP + j];
  }
  Wc32[idx] = s;
}

// Pre-swizzle c-GEMM B operand into WMMA fragment order: [kt*NT_C+nt][lane][e]
__global__ void k_swz_wc(const float* __restrict__ Wc32, _Float16* __restrict__ dst) {
  int idx = blockIdx.x * blockDim.x + threadIdx.x;
  if (idx >= KT_C * NT_C * 512) return;
  int e = idx & 15, lane = (idx >> 4) & 31;
  int tile = idx >> 9;
  int nt = tile % NT_C, kt = tile / NT_C;
  int hi = lane >> 4, ln = lane & 15;
  int k = kt * 32 + ((e < 8) ? e : e + 8) + hi * 8;  // B-fragment element e -> K = (e<8?e:e+8) + hi*8
  int n = nt * 16 + ln;
  dst[idx] = (_Float16)Wc32[k * NP + n];
}

// Pre-swizzle y/resid-GEMM B operand (basis^T, padded) into fragment order
__global__ void k_swz_bt(const float* __restrict__ basis, _Float16* __restrict__ dst) {
  int idx = blockIdx.x * blockDim.x + threadIdx.x;
  if (idx >= KT_Y * NT_Y * 512) return;
  int e = idx & 15, lane = (idx >> 4) & 31;
  int tile = idx >> 9;
  int nt = tile % NT_Y, kt = tile / NT_Y;
  int hi = lane >> 4, ln = lane & 15;
  int j = kt * 32 + ((e < 8) ? e : e + 8) + hi * 8;  // slot index (K dim), table order
  int h = nt * 16 + ln;
  dst[idx] = (j < NTOT) ? (_Float16)basis[h * NTOT + j] : (_Float16)0.f;
}

__global__ void k_zerof(float* __restrict__ p, int n) {
  int idx = blockIdx.x * blockDim.x + threadIdx.x;
  if (idx < n) p[idx] = 0.f;
}

__global__ void k_initstate(const float* __restrict__ tre, const float* __restrict__ tim,
                            float* __restrict__ tape_re, float* __restrict__ tape_im,
                            int* __restrict__ active, int* __restrict__ life,
                            int* __restrict__ pcnt, int* __restrict__ ptrs) {
  __shared__ float red[256];
  float s = tre[threadIdx.x] * tre[threadIdx.x] + tim[threadIdx.x] * tim[threadIdx.x];
  red[threadIdx.x] = s;
  __syncthreads();
  for (int off = 128; off > 0; off >>= 1) {
    if (threadIdx.x < off) red[threadIdx.x] += red[threadIdx.x + off];
    __syncthreads();
  }
  float inv = 1.f / fmaxf(sqrtf(red[0]), 1e-8f);
  for (int idx = threadIdx.x; idx < BATCH * NP; idx += 256) {
    int n = idx % NP;
    int a = (n < MBASE) ? 1 : 0;
    tape_re[idx] = a ? tre[n] * inv : 0.f;
    tape_im[idx] = a ? tim[n] * inv : 0.f;
    active[idx] = a; life[idx] = 0; pcnt[idx] = 0;
  }
  if (threadIdx.x < 2 * BATCH) ptrs[threadIdx.x] = 0;
}

// ------------------------------------------------------------- per step ----

// corr EMA: corr = 0.95*corr + 0.05 * tape tape^H  (base 256x256 block, per batch)
__global__ __launch_bounds__(256) void k_corr(const float* __restrict__ tre,
                                              const float* __restrict__ tim,
                                              float* __restrict__ cre,
                                              float* __restrict__ cim) {
  int b = blockIdx.x;
  int wave = threadIdx.x >> 5, lane = threadIdx.x & 31;
  int hi = lane >> 4, ln = lane & 15;
  const float* tr = tre + b * NP;
  const float* ti = tim + b * NP;
  float* CR = cre + (size_t)b * MBASE * MBASE;
  float* CI = cim + (size_t)b * MBASE * MBASE;
  for (int i = 0; i < 32; ++i) {
    int tt = wave * 32 + i;
    int pt = tt >> 4, qt = tt & 15;
    v16h bf = {}; v16h ar = {}; v16h ai = {};
    if (hi == 0) {                      // only K=0,1 are nonzero -> lanes<16
      int q = qt * 16 + ln;
      bf[0] = (_Float16)tr[q]; bf[1] = (_Float16)ti[q];
      int p = pt * 16 + ln;
      float pr = tr[p] * 0.05f, pi = ti[p] * 0.05f;
      ar[0] = (_Float16)pr;  ar[1] = (_Float16)pi;       // re: pr*qr + pi*qi
      ai[0] = (_Float16)pi;  ai[1] = (_Float16)(-pr);    // im: pi*qr - pr*qi
    }
    v8f cfr, cfi;
#pragma unroll
    for (int r = 0; r < 8; ++r) {
      int row = pt * 16 + r + hi * 8, col = qt * 16 + ln;
      cfr[r] = CR[row * MBASE + col] * 0.95f;
      cfi[r] = CI[row * MBASE + col] * 0.95f;
    }
    cfr = wmma16(ar, bf, cfr);
    cfi = wmma16(ai, bf, cfi);
#pragma unroll
    for (int r = 0; r < 8; ++r) {
      int row = pt * 16 + r + hi * 8, col = qt * 16 + ln;
      CR[row * MBASE + col] = cfr[r];
      CI[row * MBASE + col] = cfi[r];
    }
  }
}

__global__ __launch_bounds__(256) void k_argmax(const float* __restrict__ cre,
                                                const float* __restrict__ cim,
                                                int* __restrict__ mP, int* __restrict__ mQ,
                                                float* __restrict__ mV) {
  __shared__ float sv[256];
  __shared__ int   si[256];
  int b = blockIdx.x;
  const float* CR = cre + (size_t)b * MBASE * MBASE;
  const float* CI = cim + (size_t)b * MBASE * MBASE;
  float best = -1.f; int bi = 0;
  for (int idx = threadIdx.x; idx < MBASE * MBASE; idx += 256) {
    int p = idx >> 8, q = idx & 255;
    if (p == q) continue;
    float v = CR[idx] * CR[idx] + CI[idx] * CI[idx];
    if (v > best || (v == best && idx < bi)) { best = v; bi = idx; }
  }
  sv[threadIdx.x] = best; si[threadIdx.x] = bi;
  __syncthreads();
  for (int off = 128; off > 0; off >>= 1) {
    if (threadIdx.x < off) {
      if (sv[threadIdx.x + off] > sv[threadIdx.x] ||
          (sv[threadIdx.x + off] == sv[threadIdx.x] && si[threadIdx.x + off] < si[threadIdx.x])) {
        sv[threadIdx.x] = sv[threadIdx.x + off];
        si[threadIdx.x] = si[threadIdx.x + off];
      }
    }
    __syncthreads();
  }
  if (threadIdx.x == 0) { mP[b] = si[0] >> 8; mQ[b] = si[0] & 255; mV[b] = sv[0]; }
}

// One workgroup (8 wave32) handles all 16 batch rows for one scan step.
__global__ __launch_bounds__(256) void k_step(
    const float* __restrict__ x, int t, int do_cons,
    const _Float16* __restrict__ WcS,    // swizzled c-GEMM B fragments
    const _Float16* __restrict__ BtS,    // swizzled basis^T B fragments
    const float* __restrict__ tbr, const float* __restrict__ tbi,
    const float* __restrict__ etap, const float* __restrict__ alphap,
    float* __restrict__ tape_re, float* __restrict__ tape_im,
    int* __restrict__ active, int* __restrict__ life, int* __restrict__ pcnt,
    int* __restrict__ ptrs,
    const int* __restrict__ mP, const int* __restrict__ mQ, const float* __restrict__ mV,
    float* __restrict__ out) {
  extern __shared__ char smem[];
  float* tro = (float*)smem;             // [BATCH*NP] old tape re
  float* tio = tro + BATCH * NP;         // old tape im
  float* trn = tio + BATCH * NP;         // new tape re (later: dtape re)
  float* tin = trn + BATCH * NP;         // new tape im
  float* cls = tin + BATCH * NP;         // c (real)
  int*   act = (int*)(cls + BATCH * NP); // active flags
  float* pp     = (float*)(act + BATCH * NP);   // [256*8] resid partials
  float* novacc = pp + 256 * 8;                 // [BATCH]
  float* norms  = novacc + BATCH;               // [BATCH]

  int tid = threadIdx.x;
  int wave = tid >> 5, lane = tid & 31, hi = lane >> 4, ln = lane & 15;

  // ---- prefetch next step's token rows (global_prefetch_b8); 64-float strides
  if (t + 1 < SLEN) {
    for (int i = tid; i < BATCH * (H / 64); i += 256) {
      int b = i >> 4, seg = i & 15;
      __builtin_prefetch(x + ((size_t)b * SLEN + (t + 1)) * H + seg * 64, 0, 1);
    }
  }

  // ---- P0: load carry state
  for (int idx = tid; idx < BATCH * NP; idx += 256) {
    tro[idx] = tape_re[idx]; tio[idx] = tape_im[idx]; act[idx] = active[idx];
  }
  __syncthreads();

  // ---- P1: c = x_t @ Wc  (16 x NP), two N-tiles per A fragment
  for (int nt0 = wave; nt0 < NT_C; nt0 += 16) {
    const bool two = (nt0 + 8) < NT_C;
    v8f acc0 = {}, acc1 = {};
    for (int kt = 0; kt < KT_C; ++kt) {
      v16h af;
      const float* xb = x + ((size_t)ln * SLEN + t) * H + kt * 32 + hi * 8;
#pragma unroll
      for (int p = 0; p < 8; ++p) {                // contiguous K pairs -> b64 loads
        float2 v = *(const float2*)(xb + ((p < 4) ? 2 * p : 2 * p + 8));
        af[2 * p] = (_Float16)v.x; af[2 * p + 1] = (_Float16)v.y;
      }
      v16h b0 = *(const v16h*)(WcS + (size_t)((kt * NT_C + nt0) * 32 + lane) * 16);
      acc0 = wmma16(af, b0, acc0);
      if (two) {
        v16h b1 = *(const v16h*)(WcS + (size_t)((kt * NT_C + nt0 + 8) * 32 + lane) * 16);
        acc1 = wmma16(af, b1, acc1);
      }
    }
#pragma unroll
    for (int r = 0; r < 8; ++r) {
      cls[(r + hi * 8) * NP + nt0 * 16 + ln] = acc0[r];
      if (two) cls[(r + hi * 8) * NP + (nt0 + 8) * 16 + ln] = acc1[r];
    }
  }
  __syncthreads();

  // ---- P2 (cons only): novelty = ||x_t - c @ basis^T||  deterministic reduce
  if (do_cons) {
    float lp[8] = {0.f, 0.f, 0.f, 0.f, 0.f, 0.f, 0.f, 0.f};
    for (int nt0 = wave; nt0 < NT_Y; nt0 += 16) {
      v8f acc0 = {}, acc1 = {};
      for (int kt = 0; kt < KT_Y; ++kt) {
        v16h af;
#pragma unroll
        for (int e = 0; e < 16; ++e)
          af[e] = (_Float16)cls[ln * NP + kt * 32 + a16k(e, hi)];
        acc0 = wmma16(af, *(const v16h*)(BtS + (size_t)((kt * NT_Y + nt0) * 32 + lane) * 16), acc0);
        acc1 = wmma16(af, *(const v16h*)(BtS + (size_t)((kt * NT_Y + nt0 + 8) * 32 + lane) * 16), acc1);
      }
#pragma unroll
      for (int r = 0; r < 8; ++r) {
        int bb = r + hi * 8;
        float rv0 = x[((size_t)bb * SLEN + t) * H + nt0 * 16 + ln] - acc0[r];
        float rv1 = x[((size_t)bb * SLEN + t) * H + (nt0 + 8) * 16 + ln] - acc1[r];
        lp[r] += rv0 * rv0 + rv1 * rv1;
      }
    }
#pragma unroll
    for (int r = 0; r < 8; ++r) pp[tid * 8 + r] = lp[r];
    __syncthreads();
    if (tid < BATCH) {
      int b = tid, hb = b >> 3, rb = b & 7;
      float s = 0.f;
      for (int w = 0; w < 8; ++w)
        for (int l = 0; l < 16; ++l) s += pp[(w * 32 + hb * 16 + l) * 8 + rb];
      novacc[b] = s;
    }
    __syncthreads();
  }

  // ---- P3: impression update + transient decay/expiry
  float eta = etap[0];
  for (int idx = tid; idx < BATCH * NP; idx += 256) {
    int n = idx % NP;
    float r1, i1;
    if (n < NTOT) {
      float cr = cls[idx], to_r = tro[idx], to_i = tio[idx];
      float res = to_r * cr;
      r1 = to_r + eta * cr - 0.4f * res * to_i + tbr[n];
      i1 = to_i + 0.4f * res * to_r + tbi[n];
      if (act[idx] && n >= MBASE && n < MBASE + 8) {
        int lf = life[idx] - 1;
        life[idx] = lf;
        r1 *= 0.9f; i1 *= 0.9f;
        if (lf <= 0) { r1 = 0.f; i1 = 0.f; act[idx] = 0; }
      }
    } else { r1 = 0.f; i1 = 0.f; act[idx] = 0; }
    trn[idx] = r1; tin[idx] = i1;
  }
  __syncthreads();

  // ---- P4: top-2 resonance + transient bind (one thread per batch)
  if (tid < BATCH) {
    int b = tid;
    float v0 = -1e30f, v1 = -1e30f; int i0 = 0, i1 = 0;
    for (int n = 0; n < MBASE; ++n) {
      float v = tro[b * NP + n] * cls[b * NP + n];
      if (v > v0)      { v1 = v0; i1 = i0; v0 = v; i0 = n; }
      else if (v > v1) { v1 = v; i1 = n; }
    }
    if (v0 * v1 > 0.f) {
      int ptr = ptrs[b];
      int slot = MBASE + (ptr & 7);
      float ar = trn[b * NP + i0], ai2 = tin[b * NP + i0];
      float br = trn[b * NP + i1], bi2 = tin[b * NP + i1];
      trn[b * NP + slot] = 0.05f * (ar * br - ai2 * bi2);
      tin[b * NP + slot] = 0.05f * (ar * bi2 + ai2 * br);
      act[b * NP + slot] = 1;
      life[b * NP + slot] = 5;
      ptrs[b] = ptr + 1;
    }
  }
  __syncthreads();

  // ---- P5 (cons only): prune / merge / seed
  if (do_cons && tid < BATCH) {
    int b = tid;
    for (int n = MBASE; n < NTOT; ++n) {   // prune aux slots
      int idx = b * NP + n;
      float mag2 = trn[idx] * trn[idx] + tin[idx] * tin[idx];
      int below = (act[idx] && mag2 < 2.25e-4f) ? 1 : 0;
      int pc = below ? pcnt[idx] + 1 : 0;
      pcnt[idx] = pc;
      if (pc >= 2) { trn[idx] = 0.f; tin[idx] = 0.f; act[idx] = 0; }
    }
    int do_merge = mV[b] > 0.16f;          // |corr| > 0.4  (stored squared)
    int p = mP[b], q = mQ[b];
    int ps = ptrs[BATCH + b];
    int sslot = MBASE + 8 + (ps & 7);
    float mvr = trn[b * NP + p] + trn[b * NP + q];
    float mvi = tin[b * NP + p] + tin[b * NP + q];
    if (do_merge) {
      trn[b * NP + p] *= 0.85f; tin[b * NP + p] *= 0.85f;
      trn[b * NP + q] *= 0.85f; tin[b * NP + q] *= 0.85f;
    }
    int do_seed = (!do_merge) && (novacc[b] > 6.5536f);  // nov > 0.08
    if (do_merge)      { trn[b * NP + sslot] = mvr * 0.15f; tin[b * NP + sslot] = mvi * 0.15f; }
    else if (do_seed)  { trn[b * NP + sslot] = 0.05f;       tin[b * NP + sslot] = 0.f; }
    if (do_merge || do_seed) { act[b * NP + sslot] = 1; ptrs[BATCH + b] = ps + 1; }
  }
  __syncthreads();

  // ---- P6: hypersphere renorm; write state; form dtape (real) in trn
  if (tid < BATCH) {
    int b = tid; float s = 0.f;
    for (int n = 0; n < NP; ++n) {
      int idx = b * NP + n;
      if (act[idx]) s += trn[idx] * trn[idx] + tin[idx] * tin[idx];
    }
    norms[b] = 1.f / fmaxf(sqrtf(s), 1e-8f);
  }
  __syncthreads();
  for (int idx = tid; idx < BATCH * NP; idx += 256) {
    int b = idx / NP;
    float sc = act[idx] ? norms[b] : 0.f;
    float nr = trn[idx] * sc, ni = tin[idx] * sc;
    tape_re[idx] = nr; tape_im[idx] = ni;
    active[idx] = act[idx];
    trn[idx] = nr - tro[idx];     // dtape real part
  }
  __syncthreads();

  // ---- P7: y = x_t + gate * dtape @ basis^T   (16 x H)
  float gate = 1.f / (1.f + __expf(-alphap[0]));
  for (int nt0 = wave; nt0 < NT_Y; nt0 += 16) {
    v8f acc0 = {}, acc1 = {};
    for (int kt = 0; kt < KT_Y; ++kt) {
      v16h af;
#pragma unroll
      for (int e = 0; e < 16; ++e)
        af[e] = (_Float16)trn[ln * NP + kt * 32 + a16k(e, hi)];
      acc0 = wmma16(af, *(const v16h*)(BtS + (size_t)((kt * NT_Y + nt0) * 32 + lane) * 16), acc0);
      acc1 = wmma16(af, *(const v16h*)(BtS + (size_t)((kt * NT_Y + nt0 + 8) * 32 + lane) * 16), acc1);
    }
#pragma unroll
    for (int r = 0; r < 8; ++r) {
      int bb = r + hi * 8;
      size_t xi0 = ((size_t)bb * SLEN + t) * H + nt0 * 16 + ln;
      size_t xi1 = ((size_t)bb * SLEN + t) * H + (nt0 + 8) * 16 + ln;
      out[xi0] = x[xi0] + gate * acc0[r];
      out[xi1] = x[xi1] + gate * acc1[r];
    }
  }
}

// ------------------------------------------------------------------ host ----

extern "C" void kernel_launch(void* const* d_in, const int* in_sizes, int n_in,
                              void* d_out, int out_size, void* d_ws, size_t ws_size,
                              hipStream_t stream) {
  (void)in_sizes; (void)n_in; (void)out_size; (void)ws_size;
  const float* x      = (const float*)d_in[0];
  const float* tre    = (const float*)d_in[1];
  const float* tim    = (const float*)d_in[2];
  const float* tbr    = (const float*)d_in[3];
  const float* tbi    = (const float*)d_in[4];
  const float* leak   = (const float*)d_in[5];
  const float* basis  = (const float*)d_in[6];
  const float* etap   = (const float*)d_in[7];
  const float* alphap = (const float*)d_in[8];
  float* out = (float*)d_out;

  char* ws = (char*)d_ws;
  size_t off = 0;
  auto alloc = [&](size_t bytes) -> void* {
    void* p = ws + off;
    off = (off + bytes + 255) & ~(size_t)255;
    return p;
  };
  float* A     = (float*)alloc((size_t)NP * NP * 4);
  float* Xb    = (float*)alloc((size_t)NP * NP * 4);
  float* Tb    = (float*)alloc((size_t)NP * NP * 4);
  float* X2    = (float*)alloc((size_t)NP * NP * 4);
  float* invs  = (float*)alloc(256);
  float* Wc32  = (float*)alloc((size_t)H * NP * 4);
  _Float16* WcS = (_Float16*)alloc((size_t)KT_C * NT_C * 512 * 2);
  _Float16* BtS = (_Float16*)alloc((size_t)KT_Y * NT_Y * 512 * 2);
  float* cre   = (float*)alloc((size_t)BATCH * MBASE * MBASE * 4);
  float* cim   = (float*)alloc((size_t)BATCH * MBASE * MBASE * 4);
  float* tape_re = (float*)alloc((size_t)BATCH * NP * 4);
  float* tape_im = (float*)alloc((size_t)BATCH * NP * 4);
  int* active  = (int*)alloc((size_t)BATCH * NP * 4);
  int* life    = (int*)alloc((size_t)BATCH * NP * 4);
  int* pcnt    = (int*)alloc((size_t)BATCH * NP * 4);
  int* ptrs    = (int*)alloc(2 * BATCH * 4);
  int* mP      = (int*)alloc(BATCH * 4);
  int* mQ      = (int*)alloc(BATCH * 4);
  float* mV    = (float*)alloc(BATCH * 4);

  const int g2 = (NP * NP + 255) / 256;
  const int gw = (H * NP + 255) / 256;

  // setup: Gram, Newton-Schulz inverse, fold coupling into weights, swizzle to f16
  k_gram<<<g2, 256, 0, stream>>>(basis, A);
  k_invscale<<<1, 256, 0, stream>>>(A, invs);
  k_eyescale<<<g2, 256, 0, stream>>>(Xb, invs);
  for (int it = 0; it < 20; ++it) {
    k_gemmNP<<<g2, 256, 0, stream>>>(A, Xb, Tb);
    k_twominus<<<g2, 256, 0, stream>>>(Tb);
    k_gemmNP<<<g2, 256, 0, stream>>>(Xb, Tb, X2);
    float* tmp = Xb; Xb = X2; X2 = tmp;
  }
  k_fold<<<gw, 256, 0, stream>>>(basis, leak, Xb, Wc32);
  k_swz_wc<<<(KT_C * NT_C * 512 + 255) / 256, 256, 0, stream>>>(Wc32, WcS);
  k_swz_bt<<<(KT_Y * NT_Y * 512 + 255) / 256, 256, 0, stream>>>(basis, BtS);
  int nc = BATCH * MBASE * MBASE;
  k_zerof<<<(nc + 255) / 256, 256, 0, stream>>>(cre, nc);
  k_zerof<<<(nc + 255) / 256, 256, 0, stream>>>(cim, nc);
  k_initstate<<<1, 256, 0, stream>>>(tre, tim, tape_re, tape_im, active, life, pcnt, ptrs);

  const size_t smem = (size_t)(6 * BATCH * NP) * 4 + (size_t)(256 * 8 + 64) * 4;

  for (int t = 0; t < SLEN; ++t) {
    int cons = ((t & 7) == 7) ? 1 : 0;
    if (cons) k_argmax<<<BATCH, 256, 0, stream>>>(cre, cim, mP, mQ, mV);
    k_step<<<1, 256, smem, stream>>>(x, t, cons, WcS, BtS, tbr, tbi, etap, alphap,
                                     tape_re, tape_im, active, life, pcnt, ptrs,
                                     mP, mQ, mV, out);
    k_corr<<<BATCH, 256, 0, stream>>>(tape_re, tape_im, cre, cim);
  }
}
